// CausalSelfAttn_14929306321499
// MI455X (gfx1250) — matrix operand order repaired and checked by reading, not compiled, not run on hardware
//
#include <hip/hip_runtime.h>

#define BATCH 2
#define SEQ 2048
#define EMB 1024
#define HEADS 16
#define HEAD_DIM 64

typedef _Float16 half_t;
typedef __attribute__((ext_vector_type(16))) _Float16 v16h;
typedef __attribute__((ext_vector_type(8)))  _Float16 v8h;
typedef __attribute__((ext_vector_type(8)))  float    v8f;

// ---------------------------------------------------------------------------
// WMMA helpers (gfx1250: V_WMMA_F32_16X16X32_F16, D = A(16x32) * B(32x16) + C)
// ---------------------------------------------------------------------------
static __device__ __forceinline__ v8f wmma_f16(v16h a, v16h b, v8f c) {
  return __builtin_amdgcn_wmma_f32_16x16x32_f16(false, a, false, b, (short)0, c,
                                                false, false);
}

// A-operand: lane holds row M = lane%16; per 32-k step it needs halfs
// {kbase+0..7, kbase+16..23}, kbase = (lane>=16 ? 8 : 0). p = row + kstep + kbase.
static __device__ __forceinline__ v16h load_a16(const half_t* p) {
  v8h lo = *(const v8h*)p;
  v8h hi = *(const v8h*)(p + 16);
  v16h r;
#pragma unroll
  for (int i = 0; i < 8; ++i) { r[i] = lo[i]; r[i + 8] = hi[i]; }
  return r;
}

// B-operand: lane holds col N = lane%16; per 32-k step it needs halfs
// {kbase+0..7, kbase+16..23}, kbase = (lane>=16 ? 8 : 0).
static __device__ __forceinline__ v16h load_b16(const half_t* p) {
  v8h lo = *(const v8h*)p;
  v8h hi = *(const v8h*)(p + 16);
  v16h r;
#pragma unroll
  for (int i = 0; i < 8; ++i) { r[i] = lo[i]; r[i + 8] = hi[i]; }
  return r;
}

// ---------------------------------------------------------------------------
__global__ void f32_to_f16_kernel(const float* __restrict__ in,
                                  half_t* __restrict__ out, int n) {
  int i = blockIdx.x * blockDim.x + threadIdx.x;
  if (i < n) out[i] = (half_t)in[i];
}

// ---------------------------------------------------------------------------
// QKV projection: qkv[m][f] = sum_e x[m][e] * W_in[f][e] + b_in[f]
// One wave per 32(M) x 64(N) tile, software-pipelined (double-buffered
// fragments so WMMAs overlap next k-step's loads). Epilogue scatters into
// head-major Q,K [B,H,S,D] and transposed V [B,H,D,S] (f16).
// ---------------------------------------------------------------------------
__global__ __launch_bounds__(32) void qkv_gemm_kernel(
    const half_t* __restrict__ xh, const half_t* __restrict__ wh,
    const float* __restrict__ bin, half_t* __restrict__ Qh,
    half_t* __restrict__ Kh, half_t* __restrict__ VTh) {
  const int lane = threadIdx.x;
  const int lrow = lane & 15;
  const bool hi = lane >= 16;
  const int nb = blockIdx.x * 64;  // feature tile base (0..3071)
  const int mb = blockIdx.y * 32;  // flattened B*S row tile base

  const half_t* a0p = xh + (size_t)(mb + lrow) * EMB + (hi ? 8 : 0);
  const half_t* a1p = a0p + (size_t)16 * EMB;
  const half_t* b0p = wh + (size_t)(nb + lrow) * EMB + (hi ? 8 : 0);
  const half_t* b1p = b0p + (size_t)16 * EMB;
  const half_t* b2p = b0p + (size_t)32 * EMB;
  const half_t* b3p = b0p + (size_t)48 * EMB;

  v8f acc[2][4] = {};
  // prologue: fragments for k = 0
  v16h a0 = load_a16(a0p), a1 = load_a16(a1p);
  v16h b0 = load_b16(b0p), b1 = load_b16(b1p);
  v16h b2 = load_b16(b2p), b3 = load_b16(b3p);

  for (int k = 32; k < EMB; k += 32) {
    // prefetch next k-step into fresh registers (overlaps with WMMAs below)
    v16h a0n = load_a16(a0p + k), a1n = load_a16(a1p + k);
    v16h b0n = load_b16(b0p + k), b1n = load_b16(b1p + k);
    v16h b2n = load_b16(b2p + k), b3n = load_b16(b3p + k);

    acc[0][0] = wmma_f16(a0, b0, acc[0][0]);
    acc[1][0] = wmma_f16(a1, b0, acc[1][0]);
    acc[0][1] = wmma_f16(a0, b1, acc[0][1]);
    acc[1][1] = wmma_f16(a1, b1, acc[1][1]);
    acc[0][2] = wmma_f16(a0, b2, acc[0][2]);
    acc[1][2] = wmma_f16(a1, b2, acc[1][2]);
    acc[0][3] = wmma_f16(a0, b3, acc[0][3]);
    acc[1][3] = wmma_f16(a1, b3, acc[1][3]);

    a0 = a0n; a1 = a1n; b0 = b0n; b1 = b1n; b2 = b2n; b3 = b3n;
  }
  // epilogue k-step
  acc[0][0] = wmma_f16(a0, b0, acc[0][0]);
  acc[1][0] = wmma_f16(a1, b0, acc[1][0]);
  acc[0][1] = wmma_f16(a0, b1, acc[0][1]);
  acc[1][1] = wmma_f16(a1, b1, acc[1][1]);
  acc[0][2] = wmma_f16(a0, b2, acc[0][2]);
  acc[1][2] = wmma_f16(a1, b2, acc[1][2]);
  acc[0][3] = wmma_f16(a0, b3, acc[0][3]);
  acc[1][3] = wmma_f16(a1, b3, acc[1][3]);

  const int bidx = mb / SEQ;  // 32-row tiles never straddle batches
#pragma unroll
  for (int mt = 0; mt < 2; ++mt) {
    const int sbase = (mb % SEQ) + mt * 16 + (hi ? 8 : 0);
#pragma unroll
    for (int t = 0; t < 4; ++t) {
      const int f = nb + t * 16 + lrow;  // fixed per lane (C-frag N = lane%16)
      const float bias = bin[f];
      const int sec = f / EMB;  // 0:q 1:k 2:v (64-wide tiles never straddle)
      const int fl = f - sec * EMB;
      const int h = fl / HEAD_DIM;
      const int d = fl % HEAD_DIM;
      if (sec == 2) {
        half_t* dst =
            VTh + (((size_t)bidx * HEADS + h) * HEAD_DIM + d) * SEQ + sbase;
#pragma unroll
        for (int r = 0; r < 8; ++r) dst[r] = (half_t)(acc[mt][t][r] + bias);
      } else {
        half_t* base = (sec == 0 ? Qh : Kh) +
                       (((size_t)bidx * HEADS + h) * SEQ + sbase) * HEAD_DIM + d;
#pragma unroll
        for (int r = 0; r < 8; ++r)
          base[(size_t)r * HEAD_DIM] = (half_t)(acc[mt][t][r] + bias);
      }
    }
  }
}

// ---------------------------------------------------------------------------
// Flash attention: one wave per (b,h, 16-query tile). 32-key chunks,
// online softmax, P transposed C->A through LDS, O accumulated in f32.
// K and V fragments are loaded together at chunk start so V latency hides
// under score WMMAs + softmax.
// ---------------------------------------------------------------------------
__global__ __launch_bounds__(32) void attn_kernel(
    const half_t* __restrict__ Qh, const half_t* __restrict__ Kh,
    const half_t* __restrict__ VTh, half_t* __restrict__ AOh) {
  __shared__ alignas(16) _Float16 Pl[16 * 32];

  const int lane = threadIdx.x;
  const int lrow = lane & 15;
  const bool hi = lane >= 16;
  const int qt = blockIdx.x;  // query tile (16 rows)
  const int bh = blockIdx.y;  // b*HEADS + h
  const int b = bh / HEADS;
  const int h = bh - b * HEADS;

  const half_t* Qrow =
      Qh + ((size_t)bh * SEQ + qt * 16 + lrow) * HEAD_DIM + (hi ? 8 : 0);
  const half_t* Kbase = Kh + (size_t)bh * SEQ * HEAD_DIM + (hi ? 8 : 0);
  const half_t* Vbase = VTh + (size_t)bh * HEAD_DIM * SEQ + (hi ? 8 : 0);

  const v16h qa0 = load_a16(Qrow);       // k-dim 0..31
  const v16h qa1 = load_a16(Qrow + 32);  // k-dim 32..63

  float mrun[8], lrun[8];
  v8f o[4] = {};
#pragma unroll
  for (int r = 0; r < 8; ++r) { mrun[r] = -1e30f; lrun[r] = 0.f; }

  const int qlim = qt * 16;
  const int nch = (qlim + 16 + 31) >> 5;  // 32-key chunks (causal)
  const float scale = 0.125f;             // 1/sqrt(64)

  for (int c = 0; c < nch; ++c) {
    const int kb = c * 32;
    // ---- issue all K and V loads for this chunk up front ----
    const half_t* k0 = Kbase + (size_t)(kb + lrow) * HEAD_DIM;
    const half_t* k1 = Kbase + (size_t)(kb + 16 + lrow) * HEAD_DIM;
    const v16h kb00 = load_b16(k0);
    const v16h kb01 = load_b16(k0 + 32);
    const v16h kb10 = load_b16(k1);
    const v16h kb11 = load_b16(k1 + 32);
    v16h vb[4];
#pragma unroll
    for (int dt = 0; dt < 4; ++dt)
      vb[dt] = load_b16(Vbase + (size_t)(dt * 16 + lrow) * SEQ + kb);

    // ---- scores S = Q K^T (two 16-col fragments, K-dim = 64) ----
    v8f s0 = {}, s1 = {};
    s0 = wmma_f16(qa0, kb00, s0);
    s0 = wmma_f16(qa1, kb01, s0);
    s1 = wmma_f16(qa0, kb10, s1);
    s1 = wmma_f16(qa1, kb11, s1);

    // ---- scale + causal mask (C-frag: lane col n fixed, 8 rows m) ----
    float p0[8], p1[8], smax[8];
    const int key0 = kb + lrow, key1 = kb + 16 + lrow;
    const int mg0 = qlim + (hi ? 8 : 0);
#pragma unroll
    for (int r = 0; r < 8; ++r) {
      float v0 = s0[r] * scale;
      float v1 = s1[r] * scale;
      const int m = mg0 + r;
      if (key0 > m) v0 = -1e30f;
      if (key1 > m) v1 = -1e30f;
      p0[r] = v0; p1[r] = v1;
      smax[r] = fmaxf(v0, v1);
    }
    // row max over n: reduce within each 16-lane half (wave32)
#pragma unroll
    for (int off = 1; off <= 8; off <<= 1)
#pragma unroll
      for (int r = 0; r < 8; ++r)
        smax[r] = fmaxf(smax[r], __shfl_xor(smax[r], off, 32));

    float alpha[8], rsum[8];
#pragma unroll
    for (int r = 0; r < 8; ++r) {
      const float nm = fmaxf(mrun[r], smax[r]);
      alpha[r] = __expf(mrun[r] - nm);
      mrun[r] = nm;
      p0[r] = __expf(p0[r] - nm);
      p1[r] = __expf(p1[r] - nm);
      rsum[r] = p0[r] + p1[r];
    }
#pragma unroll
    for (int off = 1; off <= 8; off <<= 1)
#pragma unroll
      for (int r = 0; r < 8; ++r) rsum[r] += __shfl_xor(rsum[r], off, 32);
#pragma unroll
    for (int r = 0; r < 8; ++r) {
      lrun[r] = lrun[r] * alpha[r] + rsum[r];
      o[0][r] *= alpha[r]; o[1][r] *= alpha[r];
      o[2][r] *= alpha[r]; o[3][r] *= alpha[r];
    }

    // ---- transpose P (C layout) -> A layout via LDS (single wave, in-order)
    {
      const int m = hi ? 8 : 0;
#pragma unroll
      for (int r = 0; r < 8; ++r) {
        Pl[(m + r) * 32 + lrow] = (half_t)p0[r];
        Pl[(m + r) * 32 + 16 + lrow] = (half_t)p1[r];
      }
    }
    const v16h pa = load_a16(&Pl[lrow * 32 + (hi ? 8 : 0)]);

    // ---- O += P * V (V fragments already in registers) ----
#pragma unroll
    for (int dt = 0; dt < 4; ++dt) o[dt] = wmma_f16(pa, vb[dt], o[dt]);
  }

  // ---- normalize and scatter to AO [B,S,E] (f16) ----
  float rinv[8];
#pragma unroll
  for (int r = 0; r < 8; ++r) rinv[r] = 1.0f / lrun[r];
  half_t* aorow = AOh + ((size_t)b * SEQ + qt * 16 + (hi ? 8 : 0)) * EMB +
                  h * HEAD_DIM + lrow;
#pragma unroll
  for (int dt = 0; dt < 4; ++dt)
#pragma unroll
    for (int r = 0; r < 8; ++r)
      aorow[(size_t)r * EMB + dt * 16] = (half_t)(o[dt][r] * rinv[r]);
}

// ---------------------------------------------------------------------------
// Output projection: out[m][f] = sum_e AO[m][e] * W_out[f][e] + b_out[f]
// 32(M) x 64(N) per wave, double-buffered; writes f32 directly to d_out.
// ---------------------------------------------------------------------------
__global__ __launch_bounds__(32) void out_gemm_kernel(
    const half_t* __restrict__ ah, const half_t* __restrict__ wh,
    const float* __restrict__ bout, float* __restrict__ out) {
  const int lane = threadIdx.x;
  const int lrow = lane & 15;
  const bool hi = lane >= 16;
  const int nb = blockIdx.x * 64;
  const int mb = blockIdx.y * 32;

  const half_t* a0p = ah + (size_t)(mb + lrow) * EMB + (hi ? 8 : 0);
  const half_t* a1p = a0p + (size_t)16 * EMB;
  const half_t* b0p = wh + (size_t)(nb + lrow) * EMB + (hi ? 8 : 0);
  const half_t* b1p = b0p + (size_t)16 * EMB;
  const half_t* b2p = b0p + (size_t)32 * EMB;
  const half_t* b3p = b0p + (size_t)48 * EMB;

  v8f acc[2][4] = {};
  v16h a0 = load_a16(a0p), a1 = load_a16(a1p);
  v16h b0 = load_b16(b0p), b1 = load_b16(b1p);
  v16h b2 = load_b16(b2p), b3 = load_b16(b3p);

  for (int k = 32; k < EMB; k += 32) {
    v16h a0n = load_a16(a0p + k), a1n = load_a16(a1p + k);
    v16h b0n = load_b16(b0p + k), b1n = load_b16(b1p + k);
    v16h b2n = load_b16(b2p + k), b3n = load_b16(b3p + k);

    acc[0][0] = wmma_f16(a0, b0, acc[0][0]);
    acc[1][0] = wmma_f16(a1, b0, acc[1][0]);
    acc[0][1] = wmma_f16(a0, b1, acc[0][1]);
    acc[1][1] = wmma_f16(a1, b1, acc[1][1]);
    acc[0][2] = wmma_f16(a0, b2, acc[0][2]);
    acc[1][2] = wmma_f16(a1, b2, acc[1][2]);
    acc[0][3] = wmma_f16(a0, b3, acc[0][3]);
    acc[1][3] = wmma_f16(a1, b3, acc[1][3]);

    a0 = a0n; a1 = a1n; b0 = b0n; b1 = b1n; b2 = b2n; b3 = b3n;
  }
  acc[0][0] = wmma_f16(a0, b0, acc[0][0]);
  acc[1][0] = wmma_f16(a1, b0, acc[1][0]);
  acc[0][1] = wmma_f16(a0, b1, acc[0][1]);
  acc[1][1] = wmma_f16(a1, b1, acc[1][1]);
  acc[0][2] = wmma_f16(a0, b2, acc[0][2]);
  acc[1][2] = wmma_f16(a1, b2, acc[1][2]);
  acc[0][3] = wmma_f16(a0, b3, acc[0][3]);
  acc[1][3] = wmma_f16(a1, b3, acc[1][3]);

#pragma unroll
  for (int mt = 0; mt < 2; ++mt) {
    float* orow =
        out + (size_t)(mb + mt * 16 + (hi ? 8 : 0)) * EMB + nb + lrow;
#pragma unroll
    for (int t = 0; t < 4; ++t) {
      const float bias = bout[nb + t * 16 + lrow];
#pragma unroll
      for (int r = 0; r < 8; ++r)
        orow[(size_t)r * EMB + t * 16] = acc[mt][t][r] + bias;
    }
  }
}

// ---------------------------------------------------------------------------
extern "C" void kernel_launch(void* const* d_in, const int* in_sizes, int n_in,
                              void* d_out, int out_size, void* d_ws,
                              size_t ws_size, hipStream_t stream) {
  (void)in_sizes; (void)n_in; (void)out_size; (void)ws_size;
  const float* x     = (const float*)d_in[0];  // [B,S,E]
  const float* W_in  = (const float*)d_in[1];  // [3E,E]
  const float* b_in  = (const float*)d_in[2];  // [3E]
  const float* W_out = (const float*)d_in[3];  // [E,E]
  const float* b_out = (const float*)d_in[4];  // [E]
  float* out = (float*)d_out;                  // [B,S,E]

  // Workspace carving (f16 buffers), total ~48 MB
  char* ws = (char*)d_ws;
  half_t* xh    = (half_t*)ws; ws += (size_t)BATCH * SEQ * EMB * 2;
  half_t* winh  = (half_t*)ws; ws += (size_t)3 * EMB * EMB * 2;
  half_t* wouth = (half_t*)ws; ws += (size_t)EMB * EMB * 2;
  half_t* Qh    = (half_t*)ws; ws += (size_t)BATCH * HEADS * SEQ * HEAD_DIM * 2;
  half_t* Kh    = (half_t*)ws; ws += (size_t)BATCH * HEADS * SEQ * HEAD_DIM * 2;
  half_t* VTh   = (half_t*)ws; ws += (size_t)BATCH * HEADS * HEAD_DIM * SEQ * 2;
  half_t* AOh   = (half_t*)ws; ws += (size_t)BATCH * SEQ * EMB * 2;

  const int nx = BATCH * SEQ * EMB;
  const int nwi = 3 * EMB * EMB;
  const int nwo = EMB * EMB;
  f32_to_f16_kernel<<<(nx + 255) / 256, 256, 0, stream>>>(x, xh, nx);
  f32_to_f16_kernel<<<(nwi + 255) / 256, 256, 0, stream>>>(W_in, winh, nwi);
  f32_to_f16_kernel<<<(nwo + 255) / 256, 256, 0, stream>>>(W_out, wouth, nwo);

  // QKV: M = B*S = 4096 (128 tiles of 32), N = 3E = 3072 (48 tiles of 64)
  qkv_gemm_kernel<<<dim3(3 * EMB / 64, BATCH * SEQ / 32), 32, 0, stream>>>(
      xh, winh, b_in, Qh, Kh, VTh);

  // Attention: one wave per (query tile, b*h)
  attn_kernel<<<dim3(SEQ / 16, BATCH * HEADS), 32, 0, stream>>>(Qh, Kh, VTh,
                                                                AOh);

  // Output projection: N = E = 1024 (16 tiles of 64)
  out_gemm_kernel<<<dim3(EMB / 64, BATCH * SEQ / 32), 32, 0, stream>>>(
      AOh, wouth, b_out, out);
}
